// DotProductScaledAttention_84550726189442
// MI455X (gfx1250) — matrix operand order repaired and checked by reading, not compiled, not run on hardware
//
#include <hip/hip_runtime.h>

// CDNA5 (gfx1250) flash-attention with post-softmax masking.
// - Transposed-score formulation (S' = K*Q^T): query in lane, keys in regs.
// - One-time prepack pass: K -> bf16 row-major, V -> bf16 transposed (global ws).
// - Main loop stages K/V tiles with the Tensor Data Mover (tensor_load_to_lds),
//   double-buffered LDS, completion via s_wait_tensorcnt.
// - Softmax in base-2 with scale*log2(e) folded into Q; raw v_exp_f32 via
//   __builtin_amdgcn_exp2f (no denorm-guard code).

typedef __attribute__((ext_vector_type(16))) __bf16 v16bf;
typedef __attribute__((ext_vector_type(8)))  __bf16 v8bf;
typedef __attribute__((ext_vector_type(4)))  __bf16 v4bf;
typedef __attribute__((ext_vector_type(8)))  float  v8f;
typedef __attribute__((ext_vector_type(4)))  float  v4f;
typedef __attribute__((ext_vector_type(4)))  unsigned int v4u;
typedef __attribute__((ext_vector_type(8)))  int    v8i;
typedef __attribute__((ext_vector_type(4)))  int    v4i;

#define N_CTX 2048
#define DHEAD 64
#define KCHUNK 32
#define QSCALE 0.18033688011f   // (1/sqrt(64)) * log2(e); softmax done in base 2

// ---------------- prepack: Kbf[bh][n][d]=bf16(K), Vtg[bh][d][n]=bf16(V) ----------------
__global__ __launch_bounds__(256)
void prepack_kernel(const float* __restrict__ K, const float* __restrict__ V,
                    __bf16* __restrict__ Kbf, __bf16* __restrict__ Vtg)
{
    const int band = blockIdx.x & 63;              // 32-key band
    const int bh   = blockIdx.x >> 6;
    const int n0   = band * 32;
    const int r    = threadIdx.x >> 3;             // 0..31 row in band
    const int c8   = (threadIdx.x & 7) * 8;        // 0..56 feature base
    const size_t base = (size_t)bh * N_CTX * DHEAD;

    const float* kp = K + base + (size_t)(n0 + r) * DHEAD + c8;
    v4f k0 = *(const v4f*)(kp);
    v4f k1 = *(const v4f*)(kp + 4);
    v4bf kb0, kb1;
#pragma unroll
    for (int j = 0; j < 4; ++j) { kb0[j] = (__bf16)k0[j]; kb1[j] = (__bf16)k1[j]; }
    __bf16* kout = Kbf + base + (size_t)(n0 + r) * DHEAD + c8;
    *(v4bf*)(kout)     = kb0;
    *(v4bf*)(kout + 4) = kb1;

    const float* vp = V + base + (size_t)(n0 + r) * DHEAD + c8;
    v4f v0 = *(const v4f*)(vp);
    v4f v1 = *(const v4f*)(vp + 4);
    __bf16* vout = Vtg + base;
#pragma unroll
    for (int j = 0; j < 4; ++j) {
        vout[(size_t)(c8 + j)     * N_CTX + n0 + r] = (__bf16)v0[j];
        vout[(size_t)(c8 + 4 + j) * N_CTX + n0 + r] = (__bf16)v1[j];
    }
}

// ---------------- TDM: 2D bf16 tile global->LDS with row padding ----------------
// tile_d0 = contiguous elements per row, tile_d1 = rows, stride_el = row stride,
// pad codes per D#: interval 0=2DW,1=4,2=8,3=16,4=32 ; amount 0=1DW,1=2,...
__device__ __forceinline__ void tdm_load_2d(unsigned lds_addr, const __bf16* gptr,
                                            unsigned tile_d0, unsigned tile_d1,
                                            unsigned stride_el,
                                            unsigned pad_interval, unsigned pad_amount)
{
    const unsigned long long ga = (unsigned long long)(uintptr_t)gptr;
    v4u g0;
    g0[0] = 1u;                                          // count=1, user descriptor
    g0[1] = lds_addr;                                    // LDS byte address
    g0[2] = (unsigned)ga;                                // global_addr[31:0]
    g0[3] = (unsigned)((ga >> 32) & 0x1FFFFFFu) | 0x80000000u;  // addr[56:32] | type=2
    v8i g1;
    g1[0] = (int)((1u << 16) | (1u << 20) | (pad_interval << 22) | (pad_amount << 25));
    g1[1] = (int)((tile_d0 & 0xFFFFu) << 16);            // tensor_dim0 = tile_d0 (lo16)
    g1[2] = (int)((tile_d0 >> 16) | ((tile_d1 & 0xFFFFu) << 16)); // dim0 hi | dim1 lo
    g1[3] = (int)((tile_d1 >> 16) | (tile_d0 << 16));    // dim1 hi | tile_dim0
    g1[4] = (int)(tile_d1 & 0xFFFFu);                    // tile_dim1 (tile_dim2=0)
    g1[5] = (int)stride_el;                              // tensor_dim0_stride lo32
    g1[6] = 0;
    g1[7] = 0;
    const v4i z4 = {0, 0, 0, 0};
    const v8i z8 = {0, 0, 0, 0, 0, 0, 0, 0};
    __builtin_amdgcn_tensor_load_to_lds(g0, g1, z4, z4, z8, 0);
}

// ---------------- main attention kernel ----------------
__global__ __launch_bounds__(128)
void attn_post_mask_kernel(const float* __restrict__ Q,
                           const __bf16* __restrict__ Kbf,
                           const __bf16* __restrict__ Vtg,
                           const unsigned char* __restrict__ mask,
                           float* __restrict__ O)
{
    // double-buffered LDS tiles (row padding matches the TDM pad fields)
    __shared__ __bf16 Klds[2][32][72];   // [buf][key-in-chunk][feature]  (128B+16B pad)
    __shared__ __bf16 Vt[2][64][40];     // [buf][vdim][key-in-chunk]     (64B+16B pad)

    const int tid  = threadIdx.x;
    const int wave = tid >> 5;
    const int lane = tid & 31;
    const int ln16 = lane & 15;
    const int half = lane >> 4;

    const int qtiles = N_CTX / 64;
    const int bh     = blockIdx.x / qtiles;
    const int qt     = blockIdx.x % qtiles;
    const int qbase  = qt * 64 + wave * 16;
    const int qrow_g = qbase + ln16;

    const float* Qb = Q + (size_t)bh * N_CTX * DHEAD;
    const __bf16* Kb = Kbf + (size_t)bh * N_CTX * DHEAD;
    const __bf16* Vb = Vtg + (size_t)bh * N_CTX * DHEAD;   // [d][n]
    const unsigned char* Mb = mask + (size_t)bh * N_CTX * N_CTX;
    float* Ob = O + (size_t)bh * N_CTX * DHEAD;

    // ---- Q as B operand (Q^T), pre-scaled by QSCALE; k = 32c + 16*half + j ----
    v16bf qb[2];
    {
        const float* qrow = Qb + (size_t)qrow_g * DHEAD;
#pragma unroll
        for (int c = 0; c < 2; ++c) {
            const int base = 32 * c + 16 * half;
            v4f a0 = *(const v4f*)(qrow + base);
            v4f a1 = *(const v4f*)(qrow + base + 4);
            v4f a2 = *(const v4f*)(qrow + base + 8);
            v4f a3 = *(const v4f*)(qrow + base + 12);
#pragma unroll
            for (int j = 0; j < 4; ++j) {
                qb[c][j]      = (__bf16)(a0[j] * QSCALE);
                qb[c][4 + j]  = (__bf16)(a1[j] * QSCALE);
                qb[c][8 + j]  = (__bf16)(a2[j] * QSCALE);
                qb[c][12 + j] = (__bf16)(a3[j] * QSCALE);
            }
        }
    }

    v8f accT[4];
#pragma unroll
    for (int t = 0; t < 4; ++t)
        accT[t] = (v8f){0.f,0.f,0.f,0.f,0.f,0.f,0.f,0.f};
    float mold = -__builtin_inff();
    float lrow = 0.f;

    // ---- prologue: TDM prefetch of chunk 0 into buffer 0 (wave 0 only) ----
    if (wave == 0) {
        tdm_load_2d((unsigned)(uintptr_t)&Klds[0][0][0], Kb,            64, 32, DHEAD, 4, 3);
        tdm_load_2d((unsigned)(uintptr_t)&Vt[0][0][0],   Vb,            32, 64, N_CTX, 3, 3);
    }

    for (int kb = 0; kb < N_CTX; kb += KCHUNK) {
        const int p = (kb >> 5) & 1;

        if (wave == 0) __builtin_amdgcn_s_wait_tensorcnt(0);   // buf p ready
        __syncthreads();

        // ---- A operands: K tiles; k = 32c + 16*half + j ----
        v16bf ka[2][2];
#pragma unroll
        for (int t = 0; t < 2; ++t)
#pragma unroll
            for (int c = 0; c < 2; ++c) {
                const int krow = t * 16 + ln16;
                v8bf lo = *(const v8bf*)(&Klds[p][krow][32 * c + 16 * half]);
                v8bf hi = *(const v8bf*)(&Klds[p][krow][32 * c + 16 * half + 8]);
#pragma unroll
                for (int j = 0; j < 8; ++j) { ka[t][c][j] = lo[j]; ka[t][c][8 + j] = hi[j]; }
            }
        // ---- A operands: V^T tiles; k = key-in-chunk = 16*half + j ----
        v16bf va[4];
#pragma unroll
        for (int t = 0; t < 4; ++t) {
            const int vd = t * 16 + ln16;
            v8bf lo = *(const v8bf*)(&Vt[p][vd][16 * half]);
            v8bf hi = *(const v8bf*)(&Vt[p][vd][16 * half + 8]);
#pragma unroll
            for (int j = 0; j < 8; ++j) { va[t][j] = lo[j]; va[t][8 + j] = hi[j]; }
        }
        __syncthreads();   // all reads of buf p done

        // ---- TDM prefetch next chunk into buf p^1 (overlaps compute) ----
        if (wave == 0 && kb + KCHUNK < N_CTX) {
            const int kn = kb + KCHUNK;
            tdm_load_2d((unsigned)(uintptr_t)&Klds[p ^ 1][0][0], Kb + (size_t)kn * DHEAD, 64, 32, DHEAD, 4, 3);
            tdm_load_2d((unsigned)(uintptr_t)&Vt[p ^ 1][0][0],   Vb + kn,                 32, 64, N_CTX, 3, 3);
        }

        // ---- S' = K * (Q^T * QSCALE) : element (key = 16t+r+8h, query = ln16) ----
        v8f s[2];
#pragma unroll
        for (int t = 0; t < 2; ++t) {
            v8f c = (v8f){0.f,0.f,0.f,0.f,0.f,0.f,0.f,0.f};
            c = __builtin_amdgcn_wmma_f32_16x16x32_bf16(false, ka[t][0], false, qb[0], (short)0, c, false, false);
            c = __builtin_amdgcn_wmma_f32_16x16x32_bf16(false, ka[t][1], false, qb[1], (short)0, c, false, false);
            s[t] = c;
        }

        // ---- online softmax (base 2): in-thread reduce + ONE xor-16 shuffle ----
        float cmax = fmaxf(s[0][0], s[1][0]);
#pragma unroll
        for (int r = 1; r < 8; ++r)
            cmax = fmaxf(cmax, fmaxf(s[0][r], s[1][r]));
        cmax = fmaxf(cmax, __shfl_xor(cmax, 16, 32));

        const float mnew  = fmaxf(mold, cmax);
        const float alpha = __builtin_amdgcn_exp2f(mold - mnew);   // raw v_exp_f32
        mold = mnew;

        float p0[8], p1[8];
        float sum = 0.f;
#pragma unroll
        for (int r = 0; r < 8; ++r) {
            p0[r] = __builtin_amdgcn_exp2f(s[0][r] - mnew);
            p1[r] = __builtin_amdgcn_exp2f(s[1][r] - mnew);
            sum += p0[r] + p1[r];
        }
        sum += __shfl_xor(sum, 16, 32);       // full UNMASKED row sum
        lrow = lrow * alpha + sum;
#pragma unroll
        for (int t = 0; t < 4; ++t) accT[t] *= alpha;

        // ---- POST-softmax mask: 2 contiguous 8-byte loads of this query's row ----
        const unsigned char* mp = Mb + (size_t)qrow_g * N_CTX + kb + 8 * half;
        const unsigned long long mt0 = *(const unsigned long long*)(mp);
        const unsigned long long mt1 = *(const unsigned long long*)(mp + 16);
#pragma unroll
        for (int r = 0; r < 8; ++r) {
            if ((mt0 >> (8 * r)) & 0xffull) p0[r] = 0.f;
            if ((mt1 >> (8 * r)) & 0xffull) p1[r] = 0.f;
        }

        // ---- P^T B-operand: col = query, k = 16*half + j; one xor-16 half-swap ----
        float recv[8];
#pragma unroll
        for (int r = 0; r < 8; ++r) {
            const float send = half ? p0[r] : p1[r];
            recv[r] = __shfl_xor(send, 16, 32);
        }
        v16bf pb;
#pragma unroll
        for (int j = 0; j < 8; ++j) {
            pb[j]     = (__bf16)(half ? recv[j] : p0[j]);
            pb[8 + j] = (__bf16)(half ? p1[j]   : recv[j]);
        }

        // ---- O^T += V^T * P^T ----
#pragma unroll
        for (int t = 0; t < 4; ++t)
            accT[t] = __builtin_amdgcn_wmma_f32_16x16x32_bf16(false, va[t], false, pb, (short)0, accT[t], false, false);
    }

    // ---- epilogue ----
    const float inv = 1.0f / lrow;
    float* orow = Ob + (size_t)qrow_g * DHEAD;
#pragma unroll
    for (int t = 0; t < 4; ++t) {
        const int base = t * 16 + 8 * half;
        v4f o0, o1;
#pragma unroll
        for (int j = 0; j < 4; ++j) { o0[j] = accT[t][j] * inv; o1[j] = accT[t][4 + j] * inv; }
        *(v4f*)(orow + base)     = o0;
        *(v4f*)(orow + base + 4) = o1;
    }
}

extern "C" void kernel_launch(void* const* d_in, const int* in_sizes, int n_in,
                              void* d_out, int out_size, void* d_ws, size_t ws_size,
                              hipStream_t stream) {
    const float* Q = (const float*)d_in[0];
    const float* K = (const float*)d_in[1];
    const float* V = (const float*)d_in[2];
    const unsigned char* mask = (const unsigned char*)d_in[3];  // jnp.bool_ -> 1 byte
    float* O = (float*)d_out;

    const int BH = 4 * 16;
    __bf16* Kbf = (__bf16*)d_ws;                               // 16.78 MB
    __bf16* Vtg = Kbf + (size_t)BH * N_CTX * DHEAD;            // 16.78 MB

    prepack_kernel<<<dim3(BH * 64), dim3(256), 0, stream>>>(K, V, Kbf, Vtg);

    const int qtiles = N_CTX / 64;
    attn_post_mask_kernel<<<dim3(BH * qtiles), dim3(128), 0, stream>>>(Q, Kbf, Vtg, mask, O);
}
